// SelfAtt_12171937317234
// MI455X (gfx1250) — matrix operand order repaired and checked by reading, not compiled, not run on hardware
//
#include <hip/hip_runtime.h>
#include <hip/hip_bf16.h>

typedef __attribute__((ext_vector_type(16))) _Float16 v16h;
typedef __attribute__((ext_vector_type(8)))  float    v8f;

#define NN   12288
#define BB   1024
#define DIN  256
#define DHID 256
#define DK   64

// ---------------------------------------------------------------------------
// f32 -> f16 weight conversion
// ---------------------------------------------------------------------------
__global__ __launch_bounds__(256) void cvt_f32_f16(const float* __restrict__ s,
                                                   _Float16* __restrict__ d, int n) {
    int i = blockIdx.x * blockDim.x + threadIdx.x;
    if (i < n) d[i] = (_Float16)s[i];
}

// ---------------------------------------------------------------------------
// Fused 2-layer MLP for q and k, 16 nodes per block, hidden kept in LDS.
//   layer1: [16,256] x W1^T[256,256] + b1, ReLU  -> LDS f16
//   layer2: [16,256] x W2^T[256,64]  + b2        -> global f32 q/k
// WMMA tile layouts (wave32):
//   A 16x32 f16: lane L<16 holds row L, K 0..15 contiguous; lane L+16: K 16..31
//   B 32x16 f16: lane L<16 holds col L, K 0..15 contiguous; lane L+16: K 16..31
//   C/D 16x16 f32: lane holds col lane&15; VGPR r = row r (+8 for lanes>=16)
// ---------------------------------------------------------------------------
__global__ __launch_bounds__(512) void mlp_fused(
    const float* __restrict__ x,
    const _Float16* __restrict__ qw1h, const float* __restrict__ qb1,
    const _Float16* __restrict__ qw2h, const float* __restrict__ qb2,
    const _Float16* __restrict__ kw1h, const float* __restrict__ kb1,
    const _Float16* __restrict__ kw2h, const float* __restrict__ kb2,
    float* __restrict__ qout, float* __restrict__ kout)
{
    __shared__ __align__(16) _Float16 hid[2][16][264];   // padded stride: 528B % 256B != 0

    const int wid  = threadIdx.x >> 5;   // wave id 0..15
    const int lane = threadIdx.x & 31;
    const int col  = lane & 15;          // C column / A row / B column
    const int hi   = lane >> 4;          // lane-half select
    const int row0 = blockIdx.x * 16;

    // ---------------- Phase 1: hidden layer (32 tiles of 16x16, 2 per wave) --
    for (int t = wid; t < 32; t += 16) {
        const int sel = t >> 4;          // 0 = q, 1 = k
        const int ct  = t & 15;          // hidden column tile
        const _Float16* w1 = sel ? kw1h : qw1h;
        const float*    b1 = sel ? kb1  : qb1;

        v8f acc = {};
        const int xrow = row0 + col;
        const int wrow = ct * 16 + col;
        for (int kt = 0; kt < 8; ++kt) {
            const int ka = kt * 32 + hi * 16;
            const float* xp = x + (size_t)xrow * DIN + ka;
            v16h a;
            #pragma unroll
            for (int e = 0; e < 8; ++e) a[e]     = (_Float16)xp[e];
            #pragma unroll
            for (int e = 0; e < 8; ++e) a[8 + e] = (_Float16)xp[8 + e];
            const int kb = kt * 32 + hi * 16;
            v16h b = *(const v16h*)(w1 + (size_t)wrow * DIN + kb);
            acc = __builtin_amdgcn_wmma_f32_16x16x32_f16(
                      false, a, false, b, (short)0, acc, false, false);
        }
        const float bias = b1[ct * 16 + col];
        #pragma unroll
        for (int r = 0; r < 8; ++r) {
            float v = acc[r] + bias;
            v = v > 0.0f ? v : 0.0f;                 // ReLU
            hid[sel][r + 8 * hi][ct * 16 + col] = (_Float16)v;
        }
    }
    __syncthreads();

    // ---------------- Phase 2: output layer (8 tiles of 16x16) --------------
    if (wid < 8) {
        const int sel = wid >> 2;        // 0 = q, 1 = k
        const int ct  = wid & 3;         // output column tile (64/16)
        const _Float16* w2 = sel ? kw2h : qw2h;
        const float*    b2 = sel ? kb2  : qb2;
        float*          out = sel ? kout : qout;

        v8f acc = {};
        const int orow = ct * 16 + col;  // output unit
        for (int kt = 0; kt < 8; ++kt) {
            const int ka = kt * 32 + hi * 16;
            const _Float16* hp = &hid[sel][col][ka];
            v16h a;
            #pragma unroll
            for (int e = 0; e < 8; ++e) a[e]     = hp[e];
            #pragma unroll
            for (int e = 0; e < 8; ++e) a[8 + e] = hp[8 + e];
            const int kb = kt * 32 + hi * 16;
            v16h b = *(const v16h*)(w2 + (size_t)orow * DHID + kb);
            acc = __builtin_amdgcn_wmma_f32_16x16x32_f16(
                      false, a, false, b, (short)0, acc, false, false);
        }
        const float bias = b2[orow];
        #pragma unroll
        for (int r = 0; r < 8; ++r) {
            const int node = row0 + r + 8 * hi;
            out[(size_t)node * DK + orow] = acc[r] + bias;
        }
    }
}

// ---------------------------------------------------------------------------
// w_i = sum_{j in graph(i)} exp(q_i . k_j / 64)   (block-diagonal attention)
// ---------------------------------------------------------------------------
__global__ __launch_bounds__(256) void attn_w(
    const float* __restrict__ q, const float* __restrict__ k,
    const int* __restrict__ batch, float* __restrict__ w, int n)
{
    int i = blockIdx.x * blockDim.x + threadIdx.x;
    if (i >= n) return;
    const int g = batch[i];

    // lower bound of g
    int lo = 0, up = n;
    while (lo < up) { int m = (lo + up) >> 1; if (batch[m] < g) lo = m + 1; else up = m; }
    const int start = lo;
    // upper bound of g (first index > g); i is inside the run
    lo = i; up = n;
    while (lo < up) { int m = (lo + up) >> 1; if (batch[m] <= g) lo = m + 1; else up = m; }
    const int end = lo;

    // hoist q_i into registers
    float qi[DK];
    const float* qp = q + (size_t)i * DK;
    #pragma unroll
    for (int t = 0; t < DK; ++t) qi[t] = qp[t];

    float acc = 0.0f;
    for (int j = start; j < end; ++j) {
        const float* kj = k + (size_t)j * DK;
        float d = 0.0f;
        #pragma unroll
        for (int t = 0; t < DK; ++t) d += qi[t] * kj[t];
        acc += expf(d * (1.0f / (float)DK));
    }
    w[i] = acc;
}

// ---------------------------------------------------------------------------
// out[g,d] = (sum_i x[i,d]*w_i) / (sum_i w_i + 1e-6), i in graph g
// ---------------------------------------------------------------------------
__global__ __launch_bounds__(256) void pool(
    const float* __restrict__ x, const int* __restrict__ batch,
    const float* __restrict__ w, float* __restrict__ out, int n)
{
    const int g = blockIdx.x;
    const int d = threadIdx.x;

    int lo = 0, up = n;
    while (lo < up) { int m = (lo + up) >> 1; if (batch[m] < g) lo = m + 1; else up = m; }
    const int start = lo;
    up = n;
    while (lo < up) { int m = (lo + up) >> 1; if (batch[m] < g + 1) lo = m + 1; else up = m; }
    const int end = lo;

    float acc = 0.0f, aw = 0.0f;
    for (int i = start; i < end; ++i) {
        const float wi = w[i];
        acc += x[(size_t)i * DIN + d] * wi;
        aw  += wi;
    }
    out[(size_t)g * DIN + d] = acc / (aw + 1e-6f);
}

// ---------------------------------------------------------------------------
extern "C" void kernel_launch(void* const* d_in, const int* in_sizes, int n_in,
                              void* d_out, int out_size, void* d_ws, size_t ws_size,
                              hipStream_t stream) {
    (void)in_sizes; (void)n_in; (void)out_size; (void)ws_size;

    const float* x     = (const float*)d_in[0];
    const int*   batch = (const int*)d_in[1];
    const float* q_w1  = (const float*)d_in[2];
    const float* q_b1  = (const float*)d_in[3];
    const float* q_w2  = (const float*)d_in[4];
    const float* q_b2  = (const float*)d_in[5];
    const float* k_w1  = (const float*)d_in[6];
    const float* k_b1  = (const float*)d_in[7];
    const float* k_w2  = (const float*)d_in[8];
    const float* k_b2  = (const float*)d_in[9];
    float* out = (float*)d_out;

    char* ws = (char*)d_ws;
    _Float16* qw1h = (_Float16*)(ws + 0);          // 256*256 f16 = 131072 B
    _Float16* kw1h = (_Float16*)(ws + 131072);
    _Float16* qw2h = (_Float16*)(ws + 262144);     //  64*256 f16 =  32768 B
    _Float16* kw2h = (_Float16*)(ws + 294912);
    float*    qf   = (float*)   (ws + 327680);     // 12288*64 f32 = 3145728 B
    float*    kf   = (float*)   (ws + 3473408);
    float*    wbuf = (float*)   (ws + 6619136);    // 12288 f32

    cvt_f32_f16<<<(DHID * DIN + 255) / 256, 256, 0, stream>>>(q_w1, qw1h, DHID * DIN);
    cvt_f32_f16<<<(DHID * DIN + 255) / 256, 256, 0, stream>>>(k_w1, kw1h, DHID * DIN);
    cvt_f32_f16<<<(DK   * DHID + 255) / 256, 256, 0, stream>>>(q_w2, qw2h, DK * DHID);
    cvt_f32_f16<<<(DK   * DHID + 255) / 256, 256, 0, stream>>>(k_w2, kw2h, DK * DHID);

    mlp_fused<<<NN / 16, 512, 0, stream>>>(x,
                                           qw1h, q_b1, qw2h, q_b2,
                                           kw1h, k_b1, kw2h, k_b2,
                                           qf, kf);

    attn_w<<<(NN + 255) / 256, 256, 0, stream>>>(qf, kf, batch, wbuf, NN);

    pool<<<BB, 256, 0, stream>>>(x, batch, wbuf, out, NN);
}
